// ResidualVectorQuantizer_43078521979432
// MI455X (gfx1250) — matrix operand order repaired and checked by reading, not compiled, not run on hardware
//
#include <hip/hip_runtime.h>

// ---- problem constants (from reference setup_inputs) ----
#define NQ   4
#define NK   1024
#define NC   512
#define NTOK (16 * 2048)           // B*T = 32768
#define COMMIT_PLUS_ONE 1.25f      // (1 + commitment_weight) forward loss factor

typedef __attribute__((ext_vector_type(16))) __bf16 v16bf;
typedef __attribute__((ext_vector_type(8)))  float  v8f;
typedef __attribute__((ext_vector_type(4)))  __bf16 v4bf;

// ---------- workspace layout ----------
// ws + 0      : int     mask_count
// ws + 64     : float   b2[NQ*NK]                (16 KB)
// ws + 16448  : __bf16  cb_bf16[NQ*NK*NC]        (4 MB)
#define WS_B2_OFF   64
#define WS_CBBF_OFF 16448

// ---------- output layout (bytes from d_out) ----------
#define OUT_IDX_BYTES  ((size_t)NTOK * NC * 4)
#define OUT_LOSS_BYTES (OUT_IDX_BYTES + (size_t)NTOK * NQ * 8)

// ---------- LDS layout (bytes) ----------
// residual : 8 waves x 16 tokens x 512 f32  = 262144
// B tiles  : 2 x (16 codes x 512 bf16)      =  32768
// idxs     : 8 waves x 16 int               =    512
#define LDS_B_OFF    262144
#define LDS_BTILE    (16 * NC * 2)       // 16 KB per buffer
#define LDS_IDX_OFF  (LDS_B_OFF + 2 * LDS_BTILE)
#define LDS_TOTAL    (LDS_IDX_OFF + 8 * 16 * 4)

// =====================================================================
// Kernel 0: zero the accumulators (mask count in ws, loss in d_out)
// =====================================================================
__global__ void rvq_zero_kernel(int* cnt, float* loss) {
    if (threadIdx.x == 0) {
        *cnt = 0;
        *loss = 0.0f;
    }
}

// =====================================================================
// Kernel 1: count valid mask entries (n_valid)
// =====================================================================
__global__ void rvq_mask_count_kernel(const unsigned char* mask, int n, int* cnt) {
    int i = blockIdx.x * blockDim.x + threadIdx.x;
    int v = (i < n && mask[i] != 0) ? 1 : 0;
    unsigned long long b = __ballot(v);
    if ((threadIdx.x & 31) == 0) {
        atomicAdd(cnt, (int)__popcll(b));
    }
}

// =====================================================================
// Kernel 2: convert codebooks fp32 -> bf16, compute per-code norms b2
// =====================================================================
__global__ void rvq_prep_cb_kernel(const float* __restrict__ cb,
                                   __bf16* __restrict__ cb_bf,
                                   float* __restrict__ b2) {
    int lane = threadIdx.x & 31;
    int wave = threadIdx.x >> 5;
    int code = blockIdx.x * 8 + wave;          // 0 .. NQ*NK-1
    if (code >= NQ * NK) return;

    const float* src = cb + (size_t)code * NC;
    __bf16*      dst = cb_bf + (size_t)code * NC;

    float s = 0.0f;
    for (int c = lane * 4; c < NC; c += 128) {
        float4 v = *(const float4*)(src + c);
        s += v.x * v.x + v.y * v.y + v.z * v.z + v.w * v.w;
        v4bf o;
        o[0] = (__bf16)v.x; o[1] = (__bf16)v.y;
        o[2] = (__bf16)v.z; o[3] = (__bf16)v.w;
        *(v4bf*)(dst + c) = o;
    }
    #pragma unroll
    for (int off = 16; off > 0; off >>= 1) s += __shfl_xor(s, off, 32);
    if (lane == 0) b2[code] = s;
}

// ---- async global->LDS copy of one 16 B chunk (ASYNCcnt-tracked) ----
__device__ __forceinline__ void async_copy_b128(const void* gsrc, void* ldst) {
    unsigned           laddr = (unsigned)(unsigned long long)(uintptr_t)ldst;
    unsigned long long gaddr = (unsigned long long)(uintptr_t)gsrc;
    asm volatile("global_load_async_to_lds_b128 %0, %1, off"
                 :: "v"(laddr), "v"(gaddr) : "memory");
}
__device__ __forceinline__ void wait_asynccnt0() {
    asm volatile("s_wait_asynccnt 0x0" ::: "memory");
}

// =====================================================================
// Main RVQ kernel.
// block = 256 threads (8 waves); each wave owns 16 tokens (fp32 residual
// tile in LDS). The 16-code B tiles are staged into LDS with async loads
// (double buffered) and SHARED by all 8 waves -> 8x less L2 traffic.
// Distance GEMM via v_wmma_f32_16x16x32_bf16.
// =====================================================================
__global__ __launch_bounds__(256)
void rvq_main_kernel(const float* __restrict__ x,
                     const unsigned char* __restrict__ mask,
                     const float* __restrict__ cb,          // fp32 codebooks
                     const __bf16* __restrict__ cb_bf,      // bf16 codebooks
                     const float* __restrict__ b2,          // code norms
                     const int* __restrict__ cntp,          // n_valid
                     float* __restrict__ xq_out,
                     long long* __restrict__ idx_out,
                     float* __restrict__ loss_out) {
    extern __shared__ char smemraw[];

    const int lane  = threadIdx.x & 31;
    const int wave  = threadIdx.x >> 5;
    const int tid   = threadIdx.x;
    const int khalf = lane >> 4;       // 0/1: K half-group fed by this lane
    const int mrow  = lane & 15;       // A row (token-in-tile) / B row (code-in-tile)

    float*  rw   = (float*)smemraw + (size_t)wave * 16 * NC;  // 16x512 fp32 residual
    __bf16* ldsB = (__bf16*)(smemraw + LDS_B_OFF);            // 2 x (16x512) bf16
    int*    idxs = (int*)(smemraw + LDS_IDX_OFF) + wave * 16;

    const int wtok = blockIdx.x * 128 + wave * 16;            // first token of wave

    // ---- load x into the fp32 residual tile ----
    for (int m = 0; m < 16; ++m) {
        const float* xs = x + (size_t)(wtok + m) * NC;
        float*       rd = rw + m * NC;
        for (int c = lane * 4; c < NC; c += 128)
            *(float4*)(rd + c) = *(const float4*)(xs + c);
    }

    float nv = (float)(*cntp);
    nv       = fmaxf(nv, 1.0f);
    const float scl = COMMIT_PLUS_ONE / (nv * (float)NC);
    float lossAcc = 0.0f;

    for (int q = 0; q < NQ; ++q) {
        __syncthreads();

        // ---- build bf16 A fragments (register-resident, reused across all
        //      64 N-tiles). A 16x32 in contiguous k order (B matches):
        //      halves 0..15 -> K = kk*32 + khalf*16 + {0..15}
        v16bf afrag[16];
        #pragma unroll
        for (int kk = 0; kk < 16; ++kk) {
            const float* rp = rw + mrow * NC + kk * 32 + khalf * 16;
            float4 f0 = *(const float4*)(rp);
            float4 f1 = *(const float4*)(rp + 4);
            float4 f2 = *(const float4*)(rp + 8);
            float4 f3 = *(const float4*)(rp + 12);
            v16bf a;
            a[0]  = (__bf16)f0.x; a[1]  = (__bf16)f0.y; a[2]  = (__bf16)f0.z; a[3]  = (__bf16)f0.w;
            a[4]  = (__bf16)f1.x; a[5]  = (__bf16)f1.y; a[6]  = (__bf16)f1.z; a[7]  = (__bf16)f1.w;
            a[8]  = (__bf16)f2.x; a[9]  = (__bf16)f2.y; a[10] = (__bf16)f2.z; a[11] = (__bf16)f2.w;
            a[12] = (__bf16)f3.x; a[13] = (__bf16)f3.y; a[14] = (__bf16)f3.z; a[15] = (__bf16)f3.w;
            afrag[kk] = a;
        }

        float bestv[8];
        int   besti[8];
        #pragma unroll
        for (int j = 0; j < 8; ++j) { bestv[j] = -3.4e38f; besti[j] = 0; }

        const char*  cbq_bytes = (const char*)(cb_bf + (size_t)q * NK * NC);
        const float* b2q       = b2 + q * NK;

        // ---- preload B tile 0 into LDS (async, all 256 threads) ----
        {
            const char* g = cbq_bytes + tid * 16;
            char*       l = (char*)ldsB + tid * 16;
            #pragma unroll
            for (int r = 0; r < 4; ++r)
                async_copy_b128(g + r * 4096, l + r * 4096);
            wait_asynccnt0();
            __syncthreads();
        }

        // ---- sweep the 1024 codes in 16-wide N tiles (double-buffered) ----
        for (int nt = 0; nt < 64; ++nt) {
            int cur = nt & 1;

            // async-prefetch next tile into the other buffer
            if (nt + 1 < 64) {
                const char* g = cbq_bytes + (size_t)(nt + 1) * LDS_BTILE + tid * 16;
                char*       l = (char*)ldsB + (cur ^ 1) * LDS_BTILE + tid * 16;
                #pragma unroll
                for (int r = 0; r < 4; ++r)
                    async_copy_b128(g + r * 4096, l + r * 4096);
            }

            // B 32x16 fragment from LDS: lanes 0-15 hold K 0..15,
            // lanes 16-31 hold K 16..31, contiguous along C.
            const __bf16* bt = ldsB + cur * (16 * NC) + mrow * NC + khalf * 16;

            v8f acc = {0.f, 0.f, 0.f, 0.f, 0.f, 0.f, 0.f, 0.f};
            #pragma unroll
            for (int kk = 0; kk < 16; ++kk) {
                v16bf bfrag = *(const v16bf*)(bt + kk * 32);
                acc = __builtin_amdgcn_wmma_f32_16x16x32_bf16(
                        false, afrag[kk], false, bfrag,
                        (short)0, acc, false, false);
            }

            int   col = nt * 16 + mrow;              // this lane's code column
            float b2v = b2q[col];
            // C/D layout: VGPR j -> (M = j + 8*khalf, N = lane&15)
            #pragma unroll
            for (int j = 0; j < 8; ++j) {
                float s = 2.0f * acc[j] - b2v;       // maximize == argmin dist
                if (s > bestv[j]) { bestv[j] = s; besti[j] = col; }
            }

            wait_asynccnt0();       // next tile landed
            __syncthreads();        // all waves done with 'cur', next buffer ready
        }

        // ---- argmax reduction across the 16 N-lanes of each half-wave ----
        #pragma unroll
        for (int j = 0; j < 8; ++j) {
            #pragma unroll
            for (int off = 1; off < 16; off <<= 1) {
                float ov = __shfl_xor(bestv[j], off, 32);
                int   oi = __shfl_xor(besti[j], off, 32);
                if (ov > bestv[j] || (ov == bestv[j] && oi < besti[j])) {
                    bestv[j] = ov; besti[j] = oi;
                }
            }
            if (mrow == 0) idxs[j + 8 * khalf] = besti[j];   // token m = j + 8*khalf
        }
        __syncthreads();

        // ---- emit int64 indices (-1 where unmasked) ----
        if (lane < 16) {
            int tok = wtok + lane;
            long long v = (mask[tok] != 0) ? (long long)idxs[lane] : (long long)-1;
            idx_out[(size_t)tok * NQ + q] = v;
        }

        // ---- masked residual update + exact fp32 loss ----
        const float* cbf = cb + (size_t)q * NK * NC;
        for (int m = 0; m < 16; ++m) {
            int  code = idxs[m];
            int  tok  = wtok + m;
            bool mk   = (mask[tok] != 0);
            const float* qv = cbf + (size_t)code * NC;
            float*       rp = rw + m * NC;
            for (int c = lane * 4; c < NC; c += 128) {
                float4 qq = *(const float4*)(qv + c);
                float4 rr = *(float4*)(rp + c);
                float dx = qq.x - rr.x, dy = qq.y - rr.y;
                float dz = qq.z - rr.z, dw = qq.w - rr.w;
                if (mk) {
                    lossAcc += dx * dx + dy * dy + dz * dz + dw * dw;
                    rr.x -= qq.x; rr.y -= qq.y; rr.z -= qq.z; rr.w -= qq.w;
                    *(float4*)(rp + c) = rr;
                }
            }
        }
    }

    // ---- x_q = mask ? (x - residual_final) : 0 ----
    for (int m = 0; m < 16; ++m) {
        int  tok = wtok + m;
        bool mk  = (mask[tok] != 0);
        const float* xs = x + (size_t)tok * NC;
        float*       od = xq_out + (size_t)tok * NC;
        float*       rp = rw + m * NC;
        for (int c = lane * 4; c < NC; c += 128) {
            float4 xv = *(const float4*)(xs + c);
            float4 rv = *(float4*)(rp + c);
            float4 o;
            o.x = mk ? (xv.x - rv.x) : 0.0f;
            o.y = mk ? (xv.y - rv.y) : 0.0f;
            o.z = mk ? (xv.z - rv.z) : 0.0f;
            o.w = mk ? (xv.w - rv.w) : 0.0f;
            *(float4*)(od + c) = o;
        }
    }

    // ---- loss: wave reduction then one atomic per wave ----
    #pragma unroll
    for (int off = 16; off > 0; off >>= 1) lossAcc += __shfl_xor(lossAcc, off, 32);
    if (lane == 0) atomicAdd(loss_out, lossAcc * scl);
}

// =====================================================================
extern "C" void kernel_launch(void* const* d_in, const int* in_sizes, int n_in,
                              void* d_out, int out_size, void* d_ws, size_t ws_size,
                              hipStream_t stream) {
    const float*         x    = (const float*)d_in[0];
    const unsigned char* mask = (const unsigned char*)d_in[1];
    const float*         cb   = (const float*)d_in[2];

    float*     xq_out   = (float*)d_out;
    long long* idx_out  = (long long*)((char*)d_out + OUT_IDX_BYTES);
    float*     loss_out = (float*)((char*)d_out + OUT_LOSS_BYTES);

    int*    cnt   = (int*)d_ws;
    float*  b2    = (float*)((char*)d_ws + WS_B2_OFF);
    __bf16* cb_bf = (__bf16*)((char*)d_ws + WS_CBBF_OFF);

    // 0) zero accumulators
    rvq_zero_kernel<<<1, 32, 0, stream>>>(cnt, loss_out);
    // 1) n_valid = sum(mask)
    rvq_mask_count_kernel<<<(NTOK + 255) / 256, 256, 0, stream>>>(mask, NTOK, cnt);
    // 2) codebooks -> bf16 + norms
    rvq_prep_cb_kernel<<<(NQ * NK) / 8, 256, 0, stream>>>(cb, cb_bf, b2);
    // 3) main RVQ: 128 tokens per 256-thread block, ~295 KB LDS
    rvq_main_kernel<<<NTOK / 128, 256, LDS_TOTAL, stream>>>(
        x, mask, cb, cb_bf, b2, cnt, xq_out, idx_out, loss_out);
}
